// GraphClassifierTimeConv_50483045597478
// MI455X (gfx1250) — matrix operand order repaired and checked by reading, not compiled, not run on hardware
//
#include <hip/hip_runtime.h>
#include <hip/hip_bf16.h>

// ---------------------------------------------------------------- constants
#define N_NODESC   20000
#define N_EDGESC   640000
#define N_INC      128
#define N_HIDC     256
#define N_OUTC     10
#define N_TC       32
#define KCO        101            // Chebyshev coefficients 0..100
#define N_GRAPHSC  128
#define T_CONVU    28             // t' = 0..27 feed the pool (29th conv col unused)
#define T_POOLC    14
#define NFC        (N_NODESC * N_INC)        // 2,560,000
#define FEAT_POOLC (T_POOLC * N_HIDC)        // 3584

// ---------------------------------------------------------------- types
typedef __attribute__((ext_vector_type(8)))  float          v8f;
typedef __attribute__((ext_vector_type(16))) __bf16         v16bf;
typedef __attribute__((ext_vector_type(16))) unsigned short v16u;
union BF16x16 { v16u u; v16bf b; };

__device__ __forceinline__ v8f v8f_zero() { v8f z = {}; return z; }

// f32 -> bf16 bits, round-to-nearest-even
__device__ __forceinline__ unsigned short f2bf(float f) {
  unsigned u = __float_as_uint(f);
  unsigned r = u + 0x7FFFu + ((u >> 16) & 1u);
  return (unsigned short)(r >> 16);
}

// async copy 16B/lane from global into LDS (CDNA5 ASYNCcnt path, no VGPR staging)
__device__ __forceinline__ void async_b128_to_lds(const void* gptr, void* lptr) {
  unsigned loff = (unsigned)(size_t)lptr;     // low 32 bits of generic ptr = LDS byte offset
  asm volatile("global_load_async_to_lds_b128 %0, %1, off"
               :: "v"(loff), "v"(gptr) : "memory");
}
__device__ __forceinline__ void wait_asynccnt0() {
  asm volatile("s_wait_asynccnt 0x0" ::: "memory");
}

// ---------------------------------------------------------------- utility
__global__ void k_zero(float* __restrict__ p, int n) {
  int i = (blockIdx.x * 256 + threadIdx.x) * 4;
  if (i < n) *(float4*)(p + i) = make_float4(0.f, 0.f, 0.f, 0.f);
}

// ---------------------------------------------------------------- SPMV: y += w_e * v[src]  (segment_sum by dst)
__global__ void k_spmv(const float* __restrict__ v, const int* __restrict__ ei,
                       const float* __restrict__ ew, float* __restrict__ y) {
  unsigned tid = blockIdx.x * 256u + threadIdx.x;           // E*32 threads
  int e = tid >> 5;
  int c = (tid & 31) << 2;                                  // 4 features / thread
  if (e >= N_EDGESC) return;
  int s = ei[e];
  int d = ei[N_EDGESC + e];
  float w = ew[e];
  const float4 vv = *(const float4*)(v + (size_t)s * N_INC + c);
  float* yp = y + (size_t)d * N_INC + c;
  atomicAdd(yp + 0, w * vv.x);
  atomicAdd(yp + 1, w * vv.y);
  atomicAdd(yp + 2, w * vv.z);
  atomicAdd(yp + 3, w * vv.w);
}

// ------------------------------------------- t_out = a*y + b*prev ; log bf16 copy into Tk[k]
__global__ void k_combine(const float* __restrict__ y, const float* __restrict__ prev,
                          float a, float b, float* __restrict__ out,
                          unsigned short* __restrict__ tk) {
  int i = (blockIdx.x * 256 + threadIdx.x) * 4;
  if (i >= NFC) return;
  float4 yv = *(const float4*)(y + i);
  float4 pv = *(const float4*)(prev + i);
  float4 t;
  t.x = a * yv.x + b * pv.x;  t.y = a * yv.y + b * pv.y;
  t.z = a * yv.z + b * pv.z;  t.w = a * yv.w + b * pv.w;
  *(float4*)(out + i) = t;
  ushort4 u;
  u.x = f2bf(t.x); u.y = f2bf(t.y); u.z = f2bf(t.z); u.w = f2bf(t.w);
  *(ushort4*)(tk + i) = u;
}

// ------------------------------------------- conv_w (O,I,K) f32 -> wB[o][kk] bf16, kk = k*128+i
__global__ void k_prep_w(const float* __restrict__ cw, unsigned short* __restrict__ wB) {
  int idx = blockIdx.x * 256 + threadIdx.x;                 // 256*512 = 131072
  int o = idx >> 9, kk = idx & 511;
  int k = kk >> 7, i = kk & 127;
  wB[(size_t)o * 512 + kk] = f2bf(cw[(size_t)o * 512 + i * 4 + k]);
}

// ------------------------------------------- acc[t,col] = sum_k coefs[t,k] * Tk[k,col]  (bf16 WMMA)
// GEMM: M=32 (2 tiles), K=101 padded to 128 (4 chunks of 32), N = NF columns
__global__ void k_acc_gemm(const float* __restrict__ coefs,
                           const unsigned short* __restrict__ tk,
                           unsigned short* __restrict__ accb) {
  __shared__ unsigned short cb[N_TC][128];                  // bf16 coefs, zero padded
  for (int idx = threadIdx.x; idx < N_TC * 128; idx += 256) {
    int t = idx >> 7, k = idx & 127;
    cb[t][k] = (k < KCO) ? f2bf(coefs[t * KCO + k]) : (unsigned short)0;
  }
  __syncthreads();

  int wave = threadIdx.x >> 5, lane = threadIdx.x & 31;
  int nl = lane & 15, hi = (lane >= 16) ? 1 : 0;
  int colBase = (blockIdx.x * 8 + wave) * 16;
  int col = colBase + nl;

  v8f C0 = v8f_zero(), C1 = v8f_zero();
  for (int kc = 0; kc < 128; kc += 32) {
    BF16x16 A0, A1, B;
#pragma unroll
    for (int p = 0; p < 8; ++p) {
      // A 16x32 bf16 layout (ISA 7.12.2): pair base = (p<4?0:16) + (hi?8:0) + 2*(p&3)
      int kl = ((p < 4) ? 0 : 16) + hi * 8 + 2 * (p & 3);
      unsigned ua0 = *(const unsigned*)&cb[nl][kc + kl];
      unsigned ua1 = *(const unsigned*)&cb[nl + 16][kc + kl];
      A0.u[2 * p]     = (unsigned short)ua0;
      A0.u[2 * p + 1] = (unsigned short)(ua0 >> 16);
      A1.u[2 * p]     = (unsigned short)ua1;
      A1.u[2 * p + 1] = (unsigned short)(ua1 >> 16);
      // B 32x16 bf16: same k order as A (pair base kl)
      int Kg = kc + kl;
      B.u[2 * p]     = (Kg     < KCO) ? tk[(size_t)Kg * NFC + col]       : (unsigned short)0;
      B.u[2 * p + 1] = (Kg + 1 < KCO) ? tk[(size_t)(Kg + 1) * NFC + col] : (unsigned short)0;
    }
    C0 = __builtin_amdgcn_wmma_f32_16x16x32_bf16(false, A0.b, false, B.b, (short)0, C0, false, false);
    C1 = __builtin_amdgcn_wmma_f32_16x16x32_bf16(false, A1.b, false, B.b, (short)0, C1, false, false);
  }
#pragma unroll
  for (int v = 0; v < 8; ++v) {
    int m = hi ? (v + 8) : v;                               // C/D: VGPR v -> row m (m+8 for hi lanes)
    accb[(size_t)m * NFC + col]        = f2bf(C0[v]);
    accb[(size_t)(m + 16) * NFC + col] = f2bf(C1[v]);
  }
}

// ------------------------------------------- node counts per graph
__global__ void k_count(const int* __restrict__ batch, float* __restrict__ gcnt) {
  int n = blockIdx.x * 256 + threadIdx.x;
  if (n < N_NODESC) atomicAdd(&gcnt[batch[n]], 1.0f);
}

// ------------------------------------------- fused conv(K=4) + maxpool(2) + relu + per-graph sum
// block = 16-node tile, 8 waves; wave w handles output-channel tiles w and w+8.
// acc planes are streamed into a 4-plane LDS ring with GLOBAL_LOAD_ASYNC_TO_LDS_B128.
__global__ void k_conv(const unsigned short* __restrict__ accb,
                       const unsigned short* __restrict__ wB,
                       const float* __restrict__ conv_b,
                       const int* __restrict__ batch,
                       float* __restrict__ gsum) {
  __shared__ unsigned short ldsA[4][16][128];               // 4-plane ring of acc[t][node][i]
  int nodeBase = blockIdx.x * 16;
  int wave = threadIdx.x >> 5, lane = threadIdx.x & 31;
  int nl = lane & 15, hi = (lane >= 16) ? 1 : 0;
  int oBase0 = wave * 16, oBase1 = (wave + 8) * 16;
  float bias0 = conv_b[oBase0 + nl];
  float bias1 = conv_b[oBase1 + nl];

  // per-thread 16B slice of one 4KB plane (256 threads * 16B = 4096B)
  const size_t gslice = (size_t)nodeBase * 128 + threadIdx.x * 8;   // ushort units
  const unsigned ls = threadIdx.x * 16;                             // byte offset in plane

  // preload planes t = 0,1,2 via async pipe (completion fenced before first consume)
  for (int t = 0; t < 3; ++t)
    async_b128_to_lds(accb + (size_t)t * NFC + gslice,
                      (char*)&ldsA[t][0][0] + ls);

  v8f stash0 = v8f_zero(), stash1 = v8f_zero();
  for (int t0 = 0; t0 < T_CONVU; ++t0) {
    __syncthreads();                                        // all reads of recycled slot retired
    {
      int tload = t0 + 3, slot = tload & 3;
      async_b128_to_lds(accb + (size_t)tload * NFC + gslice,
                        (char*)&ldsA[slot][0][0] + ls);
      if (t0 + 4 < N_TC)                                    // warm L2 for next plane
        __builtin_prefetch(accb + (size_t)(t0 + 4) * NFC + gslice, 0, 1);
    }
    wait_asynccnt0();                                       // this wave's async writes landed
    __syncthreads();                                        // publish plane to all waves

    v8f C0 = v8f_zero(), C1 = v8f_zero();
    for (int c = 0; c < 16; ++c) {                          // K = 512 = (i,k), kk = k*128 + i
      int plane = (t0 + (c >> 2)) & 3;                      // k = c/4
      int ib = (c & 3) * 32;                                // i chunk base
      BF16x16 A, B0, B1;
#pragma unroll
      for (int p = 0; p < 8; ++p) {
        int kl = ((p < 4) ? 0 : 16) + hi * 8 + 2 * (p & 3); // A-layout pair base
        unsigned ua = *(const unsigned*)&ldsA[plane][nl][ib + kl];
        A.u[2 * p]     = (unsigned short)ua;
        A.u[2 * p + 1] = (unsigned short)(ua >> 16);
        int kk = c * 32 + kl;                               // same k order as A
        unsigned ub0 = *(const unsigned*)&wB[(size_t)(oBase0 + nl) * 512 + kk];
        unsigned ub1 = *(const unsigned*)&wB[(size_t)(oBase1 + nl) * 512 + kk];
        B0.u[2 * p]     = (unsigned short)ub0;
        B0.u[2 * p + 1] = (unsigned short)(ub0 >> 16);
        B1.u[2 * p]     = (unsigned short)ub1;
        B1.u[2 * p + 1] = (unsigned short)(ub1 >> 16);
      }
      C0 = __builtin_amdgcn_wmma_f32_16x16x32_bf16(false, A.b, false, B0.b, (short)0, C0, false, false);
      C1 = __builtin_amdgcn_wmma_f32_16x16x32_bf16(false, A.b, false, B1.b, (short)0, C1, false, false);
    }

    if (!(t0 & 1)) { stash0 = C0; stash1 = C1; }
    else {
      int pidx = t0 >> 1;
#pragma unroll
      for (int v = 0; v < 8; ++v) {
        int node = nodeBase + (hi ? (v + 8) : v);
        int g = batch[node];
        float r0 = fmaxf(fmaxf(stash0[v], C0[v]) + bias0, 0.f);
        float r1 = fmaxf(fmaxf(stash1[v], C1[v]) + bias1, 0.f);
        atomicAdd(&gsum[(size_t)g * FEAT_POOLC + pidx * N_HIDC + oBase0 + nl], r0);
        atomicAdd(&gsum[(size_t)g * FEAT_POOLC + pidx * N_HIDC + oBase1 + nl], r1);
      }
    }
  }
}

// ------------------------------------------- mean + relu + FC + log_softmax (1 wave / graph)
__global__ void k_head(const float* __restrict__ gsum, const float* __restrict__ gcnt,
                       const float* __restrict__ fcw, const float* __restrict__ fcb,
                       float* __restrict__ out) {
  int g = blockIdx.x, lane = threadIdx.x;
  float inv = 1.0f / fmaxf(gcnt[g], 1.0f);
  float p[N_OUTC];
#pragma unroll
  for (int o = 0; o < N_OUTC; ++o) p[o] = 0.f;
  for (int j = lane; j < FEAT_POOLC; j += 32) {
    float z = fmaxf(gsum[(size_t)g * FEAT_POOLC + j] * inv, 0.f);
#pragma unroll
    for (int o = 0; o < N_OUTC; ++o) p[o] += fcw[(size_t)o * FEAT_POOLC + j] * z;
  }
#pragma unroll
  for (int o = 0; o < N_OUTC; ++o)
    for (int off = 16; off > 0; off >>= 1) p[o] += __shfl_xor(p[o], off, 32);
  if (lane == 0) {
    float l[N_OUTC], mx = -1e30f;
#pragma unroll
    for (int o = 0; o < N_OUTC; ++o) { l[o] = p[o] + fcb[o]; mx = fmaxf(mx, l[o]); }
    float s = 0.f;
#pragma unroll
    for (int o = 0; o < N_OUTC; ++o) s += __expf(l[o] - mx);
    float lse = __logf(s) + mx;
#pragma unroll
    for (int o = 0; o < N_OUTC; ++o) out[g * N_OUTC + o] = l[o] - lse;
  }
}

// ---------------------------------------------------------------- workspace layout (bytes)
#define OFF_A    ((size_t)0)                        // f32 NF
#define OFF_B    ((size_t)10240000)                 // f32 NF
#define OFF_Y    ((size_t)20480000)                 // f32 NF
#define OFF_TK   ((size_t)30720000)                 // bf16 101*NF
#define OFF_ACC  ((size_t)547840000)                // bf16 32*NF
#define OFF_WB   ((size_t)711680000)                // bf16 256*512
#define OFF_GS   ((size_t)711942144)                // f32 128*3584
#define OFF_GC   ((size_t)713777152)                // f32 128

extern "C" void kernel_launch(void* const* d_in, const int* in_sizes, int n_in,
                              void* d_out, int out_size, void* d_ws, size_t ws_size,
                              hipStream_t stream) {
  const float* x      = (const float*)d_in[0];
  const int*   ei     = (const int*)d_in[1];
  const float* ew     = (const float*)d_in[2];
  const int*   batch  = (const int*)d_in[3];
  const float* coefs  = (const float*)d_in[4];
  const float* conv_w = (const float*)d_in[5];
  const float* conv_b = (const float*)d_in[6];
  const float* fc_w   = (const float*)d_in[7];
  const float* fc_b   = (const float*)d_in[8];
  float* out = (float*)d_out;

  char* ws = (char*)d_ws;
  float*          bufA = (float*)(ws + OFF_A);
  float*          bufB = (float*)(ws + OFF_B);
  float*          y    = (float*)(ws + OFF_Y);
  unsigned short* tk   = (unsigned short*)(ws + OFF_TK);
  unsigned short* accb = (unsigned short*)(ws + OFF_ACC);
  unsigned short* wB   = (unsigned short*)(ws + OFF_WB);
  float*          gsum = (float*)(ws + OFF_GS);
  float*          gcnt = (float*)(ws + OFF_GC);

  dim3 b256(256);
  const int gNF4  = NFC / 4 / 256;          // 2500
  const int gSpmv = N_EDGESC * 32 / 256;    // 80000

  // T0 = x  (copy to f32 recursion buffer + bf16 log)
  k_combine<<<gNF4, b256, 0, stream>>>(x, x, 1.f, 0.f, bufA, tk);
  // T1 = A x
  k_zero<<<gNF4, b256, 0, stream>>>(y, NFC);
  k_spmv<<<gSpmv, b256, 0, stream>>>(bufA, ei, ew, y);
  k_combine<<<gNF4, b256, 0, stream>>>(y, bufA, 1.f, 0.f, bufB, tk + (size_t)NFC);
  // T_k = 2 A T_{k-1} - T_{k-2}
  float* prev = bufA;
  float* curr = bufB;
  for (int k = 2; k < KCO; ++k) {
    k_zero<<<gNF4, b256, 0, stream>>>(y, NFC);
    k_spmv<<<gSpmv, b256, 0, stream>>>(curr, ei, ew, y);
    k_combine<<<gNF4, b256, 0, stream>>>(y, prev, 2.f, -1.f, prev, tk + (size_t)k * NFC);
    float* t = prev; prev = curr; curr = t;
  }

  // acc = coefs x Tk   (WMMA bf16)
  k_prep_w<<<512, b256, 0, stream>>>(conv_w, wB);
  k_acc_gemm<<<NFC / 128, b256, 0, stream>>>(coefs, tk, accb);

  // per-graph accumulators
  k_zero<<<(N_GRAPHSC * FEAT_POOLC / 4 + 255) / 256, b256, 0, stream>>>(gsum, N_GRAPHSC * FEAT_POOLC);
  k_zero<<<1, b256, 0, stream>>>(gcnt, N_GRAPHSC);
  k_count<<<(N_NODESC + 255) / 256, b256, 0, stream>>>(batch, gcnt);

  // conv + pool + relu + segment sum   (WMMA bf16)
  k_conv<<<N_NODESC / 16, b256, 0, stream>>>(accb, wB, conv_b, batch, gsum);

  // head
  k_head<<<N_GRAPHSC, dim3(32), 0, stream>>>(gsum, gcnt, fc_w, fc_b, out);
}
